// SparseMultiHeadAttention_25744033972287
// MI455X (gfx1250) — matrix operand order repaired and checked by reading, not compiled, not run on hardware
//
#include <hip/hip_runtime.h>
#include <hip/hip_bf16.h>

// ---------------------------------------------------------------------------
// SparseMultiHeadAttention on MI455X (gfx1250): bf16 WMMA for all GEMMs,
// f32 accumulation, flash-attention online softmax, async LDS staging.
// ---------------------------------------------------------------------------

typedef __attribute__((ext_vector_type(16))) __bf16 v16bf;
typedef __attribute__((ext_vector_type(8)))  float  v8f;

constexpr int TN = 3072;   // tokens
constexpr int TC = 1024;   // channels
constexpr int TH = 16;     // heads
constexpr int TD = 64;     // head dim
constexpr float SCALE = 8.0f;         // sqrt(64)
constexpr float INV_SCALE = 0.125f;

// ---- WMMA fragment helpers (contiguous k = 16*half + e order used here, not the ISA table order; wave32) ----------

// 16-bit A-matrix 16x32: lane holds row M = lane&15; K index for element e:
//   half==0: K = e       -> {0..15}
//   half==1: K = e + 16  -> {16..31}
__device__ __forceinline__ int a_kmap(int e, int half) {
    return e + 16 * half;
}

// Load A fragment from a row-major row pointer (contraction stride 1).
__device__ __forceinline__ v16bf load_a_frag(const __bf16* rowptr, int half) {
    v16bf a;
#pragma unroll
    for (int e = 0; e < 16; ++e) a[e] = rowptr[a_kmap(e, half)];
    return a;
}

// B fragment when the source is laid out [n][k] (row = output col):
// lane L element e = B[k0 + 16*half + e][n] -> contiguous 16 bf16 from row n.
__device__ __forceinline__ v16bf load_bT_frag(const __bf16* p) {
    return *(const v16bf*)p;   // 32B aligned by construction
}

__device__ __forceinline__ v8f wmma_bf16(v16bf a, v16bf b, v8f c) {
    return __builtin_amdgcn_wmma_f32_16x16x32_bf16(
        /*neg_a=*/false, a, /*neg_b=*/false, b,
        /*c_mod=*/(short)0, c, /*reuse_a=*/false, /*reuse_b=*/false);
}

// ---- CDNA5 async global->LDS copy (ASYNCcnt path) --------------------------
// Per-lane 16B copy: VDST = LDS byte address (low 32 bits of the generic
// shared pointer = LDS offset), VADDR = 64-bit global address.
__device__ __forceinline__ void async_load_b128(void* lds_dst, const void* gsrc) {
    unsigned lds = (unsigned)(unsigned long long)lds_dst;
    asm volatile("global_load_async_to_lds_b128 %0, %1, off"
                 :: "v"(lds), "v"(gsrc) : "memory");
}
__device__ __forceinline__ void wait_async0() {
    asm volatile("s_wait_asynccnt 0x0" ::: "memory");
}

// ---- 1) elementwise f32 -> bf16 -------------------------------------------
__global__ __launch_bounds__(256) void cvt_bf16(const float* in, __bf16* out, int count) {
    int i = blockIdx.x * 256 + threadIdx.x;
    if (i < count) out[i] = (__bf16)in[i];
}

// ---- 1b) transpose-convert: in f32 [K][Nn] -> out bf16 [Nn][K] -------------
__global__ __launch_bounds__(256) void cvtT_bf16(const float* in, __bf16* out, int K, int Nn) {
    int i = blockIdx.x * 256 + threadIdx.x;
    if (i < K * Nn) {
        int n = i / K, k = i % K;            // coalesced writes
        out[i] = (__bf16)in[(long)k * Nn + n];
    }
}

// ---- 2) generic GEMM: C[M][Nn] = A[M][K] @ B^T[Nn][K] + bias[Nn] -----------
// block = 256 threads = 8 waves; wave w computes a 32x64 tile (2x4 frags),
// reusing each B fragment across two A fragments.
__global__ __launch_bounds__(256) void gemm_bf16(const __bf16* __restrict__ A,
                                                 const __bf16* __restrict__ BT,
                                                 const float* __restrict__ bias,
                                                 float* __restrict__ Cmat,
                                                 int M, int Nn, int K) {
    const int lane = threadIdx.x & 31;
    const int wave = threadIdx.x >> 5;
    const int half = lane >> 4;
    const int col  = lane & 15;
    const int m0 = blockIdx.y * 256 + wave * 32;
    const int n0 = blockIdx.x * 64;

    v8f acc[2][4] = {};
    const __bf16* arow0 = A + (long)(m0 + col) * K;
    const __bf16* arow1 = A + (long)(m0 + 16 + col) * K;
    for (int k0 = 0; k0 < K; k0 += 32) {
        if (k0 + 64 < K) {
            __builtin_prefetch(arow0 + k0 + 64, 0, 0);  // global_prefetch_b8
            __builtin_prefetch(arow1 + k0 + 64, 0, 0);
        }
        v16bf a0 = load_a_frag(arow0 + k0, half);
        v16bf a1 = load_a_frag(arow1 + k0, half);
#pragma unroll
        for (int nt = 0; nt < 4; ++nt) {
            v16bf b = load_bT_frag(BT + (long)(n0 + nt * 16 + col) * K + k0 + 16 * half);
            acc[0][nt] = wmma_bf16(a0, b, acc[0][nt]);
            acc[1][nt] = wmma_bf16(a1, b, acc[1][nt]);
        }
    }
    // D fragment: element r -> row m0 + mt*16 + r + 8*half, col n0 + nt*16 + col
#pragma unroll
    for (int mt = 0; mt < 2; ++mt) {
#pragma unroll
        for (int nt = 0; nt < 4; ++nt) {
            int cidx = n0 + nt * 16 + col;
            float bv = bias[cidx];
#pragma unroll
            for (int r = 0; r < 8; ++r) {
                int row = m0 + mt * 16 + r + 8 * half;
                Cmat[(long)row * Nn + cidx] = acc[mt][nt][r] + bv;
            }
        }
    }
}

// ---- 3) RMSNorm + RoPE ------------------------------------------------------
// q/k -> [H][N][D] bf16 (A-side / K-side of attention),
// v   -> [H][D][N] bf16 (transposed, so the V B-fragment is contiguous).
__global__ __launch_bounds__(256) void rms_rope(const float* __restrict__ qkv,
                                                const int* __restrict__ coords,
                                                const float* __restrict__ gq,
                                                const float* __restrict__ gk,
                                                __bf16* __restrict__ qb,
                                                __bf16* __restrict__ kb,
                                                __bf16* __restrict__ vbT) {
    __shared__ float qs[TH][TD], ks[TH][TD];
    __shared__ float cs[32], sn[32];
    __shared__ float qn[TH], kn[TH];
    const int n = blockIdx.x;
    const float* base = qkv + (long)n * (3 * TC);

    // Stage q/k rows into LDS via the CDNA5 async global->LDS path.
    {
        int t = threadIdx.x;            // 256 threads x 4 floats = 1024 floats
        float* qflat = &qs[0][0];
        float* kflat = &ks[0][0];
        async_load_b128(qflat + t * 4, base + t * 4);
        async_load_b128(kflat + t * 4, base + TC + t * 4);
        wait_async0();
    }
    if (threadIdx.x < 32) {
        int j = threadIdx.x;
        float ph = 0.0f;
        if (j < 30) {
            int axis = j / 10, fi = j % 10;
            float freq = __powf(10000.0f, -(float)fi / 10.0f);
            ph = (float)coords[n * 3 + axis] * freq;
        }
        cs[j] = __cosf(ph);
        sn[j] = __sinf(ph);
    }
    __syncthreads();
    if (threadIdx.x < TH) {
        int h = threadIdx.x;
        float sq = 0.f, sk = 0.f;
        for (int d = 0; d < TD; ++d) { sq += qs[h][d] * qs[h][d]; sk += ks[h][d] * ks[h][d]; }
        qn[h] = SCALE / fmaxf(sqrtf(sq), 1e-12f);
        kn[h] = SCALE / fmaxf(sqrtf(sk), 1e-12f);
    }
    __syncthreads();
    for (int t = threadIdx.x; t < TH * TD; t += 256) {
        int h = t >> 6, d = t & 63, j = d >> 1;
        float c = cs[j], s = sn[j];
        // q
        {
            float x0 = qs[h][2 * j]     * qn[h] * gq[h * TD + 2 * j];
            float x1 = qs[h][2 * j + 1] * qn[h] * gq[h * TD + 2 * j + 1];
            float v  = (d & 1) ? (s * x0 + c * x1) : (c * x0 - s * x1);
            qb[((long)h * TN + n) * TD + d] = (__bf16)v;
        }
        // k
        {
            float x0 = ks[h][2 * j]     * kn[h] * gk[h * TD + 2 * j];
            float x1 = ks[h][2 * j + 1] * kn[h] * gk[h * TD + 2 * j + 1];
            float v  = (d & 1) ? (s * x0 + c * x1) : (c * x0 - s * x1);
            kb[((long)h * TN + n) * TD + d] = (__bf16)v;
        }
        // v passthrough, transposed to [H][D][N]
        vbT[((long)h * TD + d) * TN + n] = (__bf16)base[2 * TC + h * TD + d];
    }
}

// ---- 4) flash attention: one wave per 16 queries per head ------------------
// grid = (TN/64, TH), block = 128 (4 independent waves).
__global__ __launch_bounds__(128) void attn(const __bf16* __restrict__ qb,
                                            const __bf16* __restrict__ kb,
                                            const __bf16* __restrict__ vbT,
                                            __bf16* __restrict__ ob) {
    const int lane = threadIdx.x & 31;
    const int wave = threadIdx.x >> 5;
    const int half = lane >> 4;
    const int col  = lane & 15;
    const int h  = blockIdx.y;
    const int q0 = blockIdx.x * 64 + wave * 16;

    const __bf16* Qh  = qb  + (long)h * TN * TD;
    const __bf16* Kh  = kb  + (long)h * TN * TD;
    const __bf16* VTh = vbT + (long)h * TD * TN;   // [D][N]

    __shared__ float  S[4][16][33];      // per-wave score tiles (padded)
    __shared__ __bf16 P[4][16][32];      // per-wave probability tiles
    __shared__ float  mrow[4][16], lrow[4][16], srow[4][16];

    // Q fragments for the two K-steps of the D=64 contraction.
    const __bf16* qrow = Qh + (long)(q0 + col) * TD;
    v16bf aq0 = load_a_frag(qrow + 0,  half);
    v16bf aq1 = load_a_frag(qrow + 32, half);

    v8f o[4] = {};
    if (lane < 16) { mrow[wave][lane] = -1e30f; lrow[wave][lane] = 0.0f; }
    __syncthreads();

    for (int k0 = 0; k0 < TN; k0 += 32) {
        // ---- S = Q K^T / sqrt(D) for this 16x32 chunk ----
#pragma unroll
        for (int nt = 0; nt < 2; ++nt) {
            v8f sfr = {};
            const __bf16* krow = Kh + (long)(k0 + nt * 16 + col) * TD;
            sfr = wmma_bf16(aq0, *(const v16bf*)(krow + 16 * half),       sfr);
            sfr = wmma_bf16(aq1, *(const v16bf*)(krow + 32 + 16 * half),  sfr);
#pragma unroll
            for (int r = 0; r < 8; ++r)
                S[wave][r + 8 * half][nt * 16 + col] = sfr[r] * INV_SCALE;
        }
        __syncthreads();
        // ---- online softmax over the chunk (lanes 0..15, one row each) ----
        if (lane < 16) {
            int row = lane;
            float m_old = mrow[wave][row], mx = m_old;
#pragma unroll
            for (int j = 0; j < 32; ++j) mx = fmaxf(mx, S[wave][row][j]);
            float scl = __expf(m_old - mx);
            float sum = 0.0f;
#pragma unroll
            for (int j = 0; j < 32; ++j) {
                float p = __expf(S[wave][row][j] - mx);
                sum += p;
                P[wave][row][j] = (__bf16)p;
            }
            mrow[wave][row] = mx;
            lrow[wave][row] = lrow[wave][row] * scl + sum;
            srow[wave][row] = scl;
        }
        __syncthreads();
        // ---- rescale accumulators, then O += P @ V ----
        v16bf ap = load_a_frag(&P[wave][col][0], half);
#pragma unroll
        for (int dt = 0; dt < 4; ++dt) {
#pragma unroll
            for (int r = 0; r < 8; ++r) o[dt][r] *= srow[wave][r + 8 * half];
            // V^T[d][n]: contiguous 16 keys from row d = dt*16+col
            v16bf bv = *(const v16bf*)(VTh + (long)(dt * 16 + col) * TN + k0 + 16 * half);
            o[dt] = wmma_bf16(ap, bv, o[dt]);
        }
        __syncthreads();   // protect S/P/srow before next chunk
    }

    // epilogue: divide by l, write bf16 attention output [n][h*64+d]
#pragma unroll
    for (int dt = 0; dt < 4; ++dt) {
#pragma unroll
        for (int r = 0; r < 8; ++r) {
            int row = q0 + r + 8 * half;
            float v = o[dt][r] / lrow[wave][r + 8 * half];
            ob[(long)row * TC + h * TD + dt * 16 + col] = (__bf16)v;
        }
    }
}

// ---------------------------------------------------------------------------
extern "C" void kernel_launch(void* const* d_in, const int* in_sizes, int n_in,
                              void* d_out, int out_size, void* d_ws, size_t ws_size,
                              hipStream_t stream) {
    const float* x      = (const float*)d_in[0];
    const int*   coords = (const int*)  d_in[1];
    const float* w_qkv  = (const float*)d_in[2];
    const float* b_qkv  = (const float*)d_in[3];
    const float* g_q    = (const float*)d_in[4];
    const float* g_k    = (const float*)d_in[5];
    const float* w_out  = (const float*)d_in[6];
    const float* b_out  = (const float*)d_in[7];
    float* out = (float*)d_out;

    // workspace layout (all chunks are multiples of 256B)
    char* ws = (char*)d_ws;
    size_t off = 0;
    __bf16* xb     = (__bf16*)(ws + off); off += (size_t)TN * TC * 2;          // 6.0 MB
    __bf16* wqkvT  = (__bf16*)(ws + off); off += (size_t)3 * TC * TC * 2;      // 6.0 MB
    float*  qkv    = (float*) (ws + off); off += (size_t)TN * 3 * TC * 4;      // 36  MB
    __bf16* qb     = (__bf16*)(ws + off); off += (size_t)TH * TN * TD * 2;     // 6.0 MB
    __bf16* kbuf   = (__bf16*)(ws + off); off += (size_t)TH * TN * TD * 2;     // 6.0 MB
    __bf16* vbufT  = (__bf16*)(ws + off); off += (size_t)TH * TD * TN * 2;     // 6.0 MB
    __bf16* ob     = (__bf16*)(ws + off); off += (size_t)TN * TC * 2;          // 6.0 MB
    __bf16* woutT  = (__bf16*)(ws + off); off += (size_t)TC * TC * 2;          // 2.0 MB
    (void)ws_size; (void)n_in; (void)in_sizes; (void)out_size;

    // 1) converts
    {
        int cnt = TN * TC;
        cvt_bf16<<<(cnt + 255) / 256, 256, 0, stream>>>(x, xb, cnt);
    }
    {
        int cnt = TC * 3 * TC;
        cvtT_bf16<<<(cnt + 255) / 256, 256, 0, stream>>>(w_qkv, wqkvT, TC, 3 * TC);
    }
    {
        int cnt = TC * TC;
        cvtT_bf16<<<(cnt + 255) / 256, 256, 0, stream>>>(w_out, woutT, TC, TC);
    }
    // 2) QKV projection: qkv[3072][3072] = xb @ wqkvT^T + b_qkv
    gemm_bf16<<<dim3(3 * TC / 64, TN / 256), 256, 0, stream>>>(
        xb, wqkvT, b_qkv, qkv, TN, 3 * TC, TC);
    // 3) RMSNorm + RoPE -> q/k bf16 [H][N][D], v bf16 [H][D][N]
    rms_rope<<<TN, 256, 0, stream>>>(qkv, coords, g_q, g_k, qb, kbuf, vbufT);
    // 4) flash attention -> ob bf16 [N][C]
    attn<<<dim3(TN / 64, TH), 128, 0, stream>>>(qb, kbuf, vbufT, ob);
    // 5) output projection: out[3072][1024] = ob @ woutT^T + b_out
    gemm_bf16<<<dim3(TC / 64, TN / 256), 256, 0, stream>>>(
        ob, woutT, b_out, out, TN, TC, TC);
}
